// WGFM_89876485636336
// MI455X (gfx1250) — matrix operand order repaired and checked by reading, not compiled, not run on hardware
//
#include <hip/hip_runtime.h>
#include <hip/hip_bf16.h>
#include <stdint.h>

// ---------------------------------------------------------------------------
// Shapes
// ---------------------------------------------------------------------------
#define BATCH 2
#define C256 256
#define H80 80
#define W80 80
#define P80 (H80*W80)          // 6400
#define H40 40
#define W40 40
#define P40 (H40*W40)          // 1600
#define EMB 64
#define KWIN 5
#define K2 (KWIN*KWIN)         // 25
#define KF 512                 // f1/f2 GEMM K
#define KR 2560                // fused r3(2304)+r1(256) GEMM K

typedef __attribute__((ext_vector_type(16))) __bf16 v16bf;
typedef __attribute__((ext_vector_type(8)))  float  v8f;

union FragAB { uint32_t u[8]; v16bf v; };

__device__ __forceinline__ uint16_t f2b(float f) {
    uint32_t u = __float_as_uint(f);
    uint32_t r = u + 0x7FFFu + ((u >> 16) & 1u);
    return (uint16_t)(r >> 16);
}
__device__ __forceinline__ float silu(float v) {
    return v / (1.0f + expf(-v));
}
__device__ __forceinline__ int reflect80(int i) {
    i = i < 0 ? -i : i;
    return i > 79 ? 158 - i : i;
}

// ---------------------------------------------------------------------------
// 1) Weight packing: fold BN gains, build fused im2col weight matrix (bf16)
// ---------------------------------------------------------------------------
__global__ void pack_weights(const float* __restrict__ f1w, const float* __restrict__ f1g,
                             const float* __restrict__ f2w, const float* __restrict__ f2g,
                             const float* __restrict__ r3w, const float* __restrict__ r3g,
                             const float* __restrict__ r1w, const float* __restrict__ r1g,
                             uint16_t* __restrict__ Wf1, uint16_t* __restrict__ Wf2,
                             uint16_t* __restrict__ Wr) {
    int i = blockIdx.x * blockDim.x + threadIdx.x;
    if (i < C256 * KF) {                 // f1 / f2: [m][k], m=out-ch
        int m = i >> 9;
        Wf1[i] = f2b(f1w[i] * f1g[m]);
        Wf2[i] = f2b(f2w[i] * f2g[m]);
    }
    if (i < C256 * KR) {                 // Wr: k = tap*256 + c (taps 0..8), then r1 block
        int m = i / KR, k = i % KR;
        float v;
        if (k < 2304) {
            int tap = k >> 8, c = k & 255;           // r3_w is [m][c][3][3]
            v = r3w[(size_t)m * 2304 + c * 9 + tap] * r3g[m];
        } else {
            v = r1w[(size_t)m * 256 + (k - 2304)] * r1g[m];
        }
        Wr[i] = f2b(v);
    }
}

// ---------------------------------------------------------------------------
// 2) Bilinear upsample x (40x40 -> 80x80), NCHW, half-pixel centers
// ---------------------------------------------------------------------------
__global__ void upsample_x(const float* __restrict__ x, float* __restrict__ x_up) {
    int idx = blockIdx.x * blockDim.x + threadIdx.x;
    if (idx >= BATCH * C256 * P80) return;
    int p = idx % P80, bc = idx / P80;
    int h = p / W80, w = p % W80;
    float sh = h * 0.5f - 0.25f, sw = w * 0.5f - 0.25f;
    int h0 = (int)floorf(sh), w0 = (int)floorf(sw);
    float th = sh - (float)h0, tw = sw - (float)w0;
    int h0c = max(h0, 0), h1c = min(h0 + 1, 39);
    int w0c = max(w0, 0), w1c = min(w0 + 1, 39);
    const float* s = x + (size_t)bc * P40;
    float v00 = s[h0c * 40 + w0c], v01 = s[h0c * 40 + w1c];
    float v10 = s[h1c * 40 + w0c], v11 = s[h1c * 40 + w1c];
    x_up[idx] = (1.f - th) * ((1.f - tw) * v00 + tw * v01)
              +        th  * ((1.f - tw) * v10 + tw * v11);
}

// ---------------------------------------------------------------------------
// 3) Haar DWT: ll = .5(a+b+c+d),  y_h = lh+hl+hh = .5(-a-b-c+3d)
// ---------------------------------------------------------------------------
__global__ void dwt_haar(const float* __restrict__ y, float* __restrict__ ll,
                         float* __restrict__ yh) {
    int idx = blockIdx.x * blockDim.x + threadIdx.x;
    if (idx >= BATCH * C256 * P40) return;
    int p = idx % P40, bc = idx / P40;
    int i = p / W40, j = p % W40;
    const float* s = y + (size_t)bc * P80;
    float a = s[(2 * i) * W80 + 2 * j],     b = s[(2 * i) * W80 + 2 * j + 1];
    float c = s[(2 * i + 1) * W80 + 2 * j], d = s[(2 * i + 1) * W80 + 2 * j + 1];
    ll[idx] = 0.5f * (a + b + c + d);
    yh[idx] = 0.5f * (-a - b - c + 3.f * d);
}

// ---------------------------------------------------------------------------
// 4) LayerNorm over 256 channels + 256->64 projection (block per pixel, LDS)
//    src NCHW [B,256,HW], out NHWC [B*HW][64]
// ---------------------------------------------------------------------------
__global__ void ln_proj(const float* __restrict__ src, int HW,
                        const float* __restrict__ g, const float* __restrict__ b_,
                        const float* __restrict__ W, const float* __restrict__ wb,
                        float* __restrict__ out) {
    __shared__ float sh[C256];
    __shared__ float rs[64], rs2[64];
    int pix = blockIdx.x;
    int bb = pix / HW, p = pix % HW;
    const float* base = src + ((size_t)bb * C256) * HW + p;
    int t = threadIdx.x;
    float vals[4], s = 0.f, s2 = 0.f;
#pragma unroll
    for (int i = 0; i < 4; ++i) {
        float v = base[(size_t)(t * 4 + i) * HW];
        vals[i] = v; s += v; s2 += v * v;
    }
    rs[t] = s; rs2[t] = s2;
    __syncthreads();
    for (int off = 32; off > 0; off >>= 1) {
        if (t < off) { rs[t] += rs[t + off]; rs2[t] += rs2[t + off]; }
        __syncthreads();
    }
    float mu = rs[0] * (1.f / 256.f);
    float var = rs2[0] * (1.f / 256.f) - mu * mu;
    float inv = rsqrtf(var + 1e-5f);
#pragma unroll
    for (int i = 0; i < 4; ++i) {
        int c = t * 4 + i;
        sh[c] = (vals[i] - mu) * inv * g[c] + b_[c];
    }
    __syncthreads();
    float acc = wb[t];
    for (int c = 0; c < C256; ++c) acc += sh[c] * W[c * EMB + t];
    out[(size_t)pix * EMB + t] = acc;
}

// ---------------------------------------------------------------------------
// 5) Cross-scale attention: q (nearest-up x2) . unfold(k)  -> softmax(25)
//    -> weighted sum of unfold(x_up) -> cat channels [0,256) (bf16 pixel-major)
// ---------------------------------------------------------------------------
__global__ void attn_kernel(const float* __restrict__ q40, const float* __restrict__ kbuf,
                            const float* __restrict__ x_up, uint16_t* __restrict__ cat) {
    int idx = blockIdx.x * blockDim.x + threadIdx.x;
    if (idx >= BATCH * P80) return;
    int b = idx / P80, p = idx % P80;
    int h = p / W80, w = p % W80;
    const float* qp = q40 + ((size_t)b * P40 + (h >> 1) * W40 + (w >> 1)) * EMB;
    float q[EMB];
#pragma unroll
    for (int e = 0; e < EMB; ++e) q[e] = qp[e];
    int off[K2];
#pragma unroll
    for (int dh = 0; dh < KWIN; ++dh) {
        int hh = reflect80(h + dh - 2);
#pragma unroll
        for (int dw = 0; dw < KWIN; ++dw)
            off[dh * KWIN + dw] = hh * W80 + reflect80(w + dw - 2);
    }
    float sim[K2], mx = -1e30f;
    const float* kb_ = kbuf + (size_t)b * P80 * EMB;
#pragma unroll
    for (int t = 0; t < K2; ++t) {
        const float* kp = kb_ + (size_t)off[t] * EMB;
        float s = 0.f;
#pragma unroll
        for (int e = 0; e < EMB; ++e) s += q[e] * kp[e];
        s *= 0.125f;                       // EMB^-0.5
        sim[t] = s; mx = fmaxf(mx, s);
    }
    float den = 0.f;
#pragma unroll
    for (int t = 0; t < K2; ++t) { sim[t] = expf(sim[t] - mx); den += sim[t]; }
    float r = 1.f / den;
#pragma unroll
    for (int t = 0; t < K2; ++t) sim[t] *= r;
    const float* xu = x_up + (size_t)b * C256 * P80;
    uint16_t* cp = cat + ((size_t)b * P80 + p) * 512;
    for (int c = 0; c < C256; ++c) {
        const float* xc = xu + (size_t)c * P80;
        float acc = 0.f;
#pragma unroll
        for (int t = 0; t < K2; ++t) acc += sim[t] * xc[off[t]];
        cp[c] = f2b(acc);
    }
}

// ---------------------------------------------------------------------------
// 6) HF mask path: com (1x1 256->64) ; mask (3x3 64->100, zero pad)
// ---------------------------------------------------------------------------
__global__ void com_kernel(const float* __restrict__ yh, const float* __restrict__ ccw,
                           const float* __restrict__ ccb, float* __restrict__ com) {
    int idx = blockIdx.x * blockDim.x + threadIdx.x;
    if (idx >= BATCH * EMB * P40) return;
    int p = idx % P40, be = idx / P40;
    int b = be / EMB, e = be % EMB;
    const float* yp = yh + (size_t)b * C256 * P40 + p;
    float acc = ccb[e];
    for (int c = 0; c < C256; ++c) acc += ccw[e * C256 + c] * yp[(size_t)c * P40];
    com[idx] = acc;
}

__global__ void mask_conv_kernel(const float* __restrict__ com, const float* __restrict__ cew,
                                 const float* __restrict__ ceb, float* __restrict__ m100) {
    int idx = blockIdx.x * blockDim.x + threadIdx.x;
    if (idx >= BATCH * 100 * P40) return;
    int p = idx % P40, bo = idx / P40;
    int b = bo / 100, o = bo % 100;
    int i = p / W40, j = p % W40;
    float acc = ceb[o];
    const float* cb = com + (size_t)b * EMB * P40;
    for (int e = 0; e < EMB; ++e) {
        const float* ce = cb + (size_t)e * P40;
        const float* wo = cew + ((size_t)o * EMB + e) * 9;
#pragma unroll
        for (int kh = 0; kh < 3; ++kh) {
            int ii = i + kh - 1;
            if (ii < 0 || ii >= H40) continue;
#pragma unroll
            for (int kw = 0; kw < 3; ++kw) {
                int jj = j + kw - 1;
                if (jj < 0 || jj >= W40) continue;
                acc += wo[kh * 3 + kw] * ce[ii * W40 + jj];
            }
        }
    }
    m100[idx] = acc;
}

// pixel-shuffle(2) + softmax(25) + x_h = y - sum_k unfold(y)*mask ->
// cat channels [256,512) (bf16 pixel-major)
__global__ void hf_kernel(const float* __restrict__ y, const float* __restrict__ m100,
                          uint16_t* __restrict__ cat) {
    int idx = blockIdx.x * blockDim.x + threadIdx.x;
    if (idx >= BATCH * P80) return;
    int b = idx / P80, p = idx % P80;
    int h = p / W80, w = p % W80;
    int di = h & 1, dj = w & 1;
    int p40 = (h >> 1) * W40 + (w >> 1);
    float m[K2], mx = -1e30f;
#pragma unroll
    for (int cc = 0; cc < K2; ++cc) {
        m[cc] = m100[((size_t)b * 100 + cc * 4 + di * 2 + dj) * P40 + p40];
        mx = fmaxf(mx, m[cc]);
    }
    float den = 0.f;
#pragma unroll
    for (int cc = 0; cc < K2; ++cc) { m[cc] = expf(m[cc] - mx); den += m[cc]; }
    float r = 1.f / den;
#pragma unroll
    for (int cc = 0; cc < K2; ++cc) m[cc] *= r;
    int off[K2];
#pragma unroll
    for (int dh = 0; dh < KWIN; ++dh) {
        int hh = reflect80(h + dh - 2);
#pragma unroll
        for (int dw = 0; dw < KWIN; ++dw)
            off[dh * KWIN + dw] = hh * W80 + reflect80(w + dw - 2);
    }
    const float* yb = y + (size_t)b * C256 * P80;
    uint16_t* cp = cat + ((size_t)b * P80 + p) * 512 + 256;
    for (int c = 0; c < C256; ++c) {
        const float* yc = yb + (size_t)c * P80;
        float hf = 0.f;
#pragma unroll
        for (int t = 0; t < K2; ++t) hf += m[t] * yc[off[t]];
        cp[c] = f2b(yc[p] - hf);
    }
}

// ---------------------------------------------------------------------------
// WMMA fragment K offsets (CDNA5 bf16 16x16x32 layouts).
// koffA: v0..3 and v4..7 each form 16B-contiguous chunks -> b128 loads.
// ---------------------------------------------------------------------------
__device__ __forceinline__ int koffA(int v, int half) {
    return (v < 4) ? (half * 8 + 2 * v) : (16 + half * 8 + 2 * (v - 4));
}
__device__ __forceinline__ int koffB(int v, int half) {
    return (v < 4) ? (half * 8 + 2 * v) : (16 + half * 8 + 2 * (v - 4));
}

#define WMMA_BF16(A, B, C) \
    __builtin_amdgcn_wmma_f32_16x16x32_bf16(false, (A).v, false, (B).v, (short)0, (C), false, false)

// ---------------------------------------------------------------------------
// 7) GEMM-F: out = silu(W[256xKF] @ cat[KF x 6400] + bias)
//    32x32 tile per wave (2 A-frags x 2 B-frags -> 4 WMMA per K-step).
//    mode 0 -> x1 (bf16, pixel-major)   mode 1 -> x2 (fp32, NCHW)
// ---------------------------------------------------------------------------
__global__ void gemm_f(const uint16_t* __restrict__ Wp, const uint16_t* __restrict__ cat,
                       const float* __restrict__ bias, uint16_t* __restrict__ x1,
                       float* __restrict__ x2, int mode) {
    int lane = threadIdx.x & 31, wave = threadIdx.x >> 5;
    int half = lane >> 4, l16 = lane & 15;
    int b = blockIdx.z;
    int mbase = blockIdx.y * 64 + (wave >> 1) * 32;
    int nbase = blockIdx.x * 64 + (wave & 1) * 32;
    const uint16_t* Arow0 = Wp + (size_t)(mbase + l16) * KF;
    const uint16_t* Arow1 = Arow0 + (size_t)16 * KF;
    const uint16_t* Brow0 = cat + ((size_t)b * P80 + nbase + l16) * 512;
    const uint16_t* Brow1 = Brow0 + (size_t)16 * 512;
    v8f a00 = {}, a01 = {}, a10 = {}, a11 = {};
    for (int k0 = 0; k0 < KF; k0 += 32) {
        __builtin_prefetch(Brow0 + k0 + 64, 0, 0);
        __builtin_prefetch(Brow1 + k0 + 64, 0, 0);
        FragAB fa0, fa1, fb0, fb1;
#pragma unroll
        for (int v = 0; v < 8; ++v) {
            int ka = k0 + koffA(v, half), kb = k0 + koffB(v, half);
            fa0.u[v] = *(const uint32_t*)(Arow0 + ka);
            fa1.u[v] = *(const uint32_t*)(Arow1 + ka);
            fb0.u[v] = *(const uint32_t*)(Brow0 + kb);
            fb1.u[v] = *(const uint32_t*)(Brow1 + kb);
        }
        a00 = WMMA_BF16(fa0, fb0, a00);
        a01 = WMMA_BF16(fa0, fb1, a01);
        a10 = WMMA_BF16(fa1, fb0, a10);
        a11 = WMMA_BF16(fa1, fb1, a11);
    }
#pragma unroll
    for (int i = 0; i < 2; ++i) {
#pragma unroll
        for (int j = 0; j < 2; ++j) {
            const v8f& acc = i == 0 ? (j == 0 ? a00 : a01) : (j == 0 ? a10 : a11);
            int n = nbase + 16 * j + l16;
#pragma unroll
            for (int v = 0; v < 8; ++v) {
                int m = mbase + 16 * i + v + 8 * half;
                float s = silu(acc[v] + bias[m]);
                if (mode == 0) x1[((size_t)b * P80 + n) * C256 + m] = f2b(s);
                else           x2[((size_t)b * C256 + m) * P80 + n] = s;
            }
        }
    }
}

// ---------------------------------------------------------------------------
// 8) GEMM-R: fused r3(3x3 zero-pad) + r1(1x1) on x1, K=2560, 32x32/wave, then
//    out = silu(acc + r3_b + r1_b) + x2      (final output, NCHW fp32)
// ---------------------------------------------------------------------------
__global__ void gemm_r(const uint16_t* __restrict__ Wr, const uint16_t* __restrict__ x1,
                       const float* __restrict__ b3, const float* __restrict__ b1,
                       const float* __restrict__ x2, float* __restrict__ out) {
    int lane = threadIdx.x & 31, wave = threadIdx.x >> 5;
    int half = lane >> 4, l16 = lane & 15;
    int b = blockIdx.z;
    int mbase = blockIdx.y * 64 + (wave >> 1) * 32;
    int nbase = blockIdx.x * 64 + (wave & 1) * 32;
    int n0 = nbase + l16, n1 = n0 + 16;
    int h0 = n0 / W80, w0 = n0 % W80;
    int h1 = n1 / W80, w1 = n1 % W80;
    const uint16_t* Arow0 = Wr + (size_t)(mbase + l16) * KR;
    const uint16_t* Arow1 = Arow0 + (size_t)16 * KR;
    const uint16_t* x1b = x1 + (size_t)b * P80 * C256;
    v8f a00 = {}, a01 = {}, a10 = {}, a11 = {};
    for (int k0 = 0; k0 < KR; k0 += 32) {
        int tap, c00;
        if (k0 < 2304) { tap = k0 >> 8; c00 = k0 & 255; }
        else           { tap = 4;       c00 = k0 - 2304; }
        int dh = tap / 3 - 1, dw = tap % 3 - 1;
        int hh0 = h0 + dh, ww0 = w0 + dw;
        int hh1 = h1 + dh, ww1 = w1 + dw;
        bool ok0 = (hh0 >= 0) && (hh0 < H80) && (ww0 >= 0) && (ww0 < W80);
        bool ok1 = (hh1 >= 0) && (hh1 < H80) && (ww1 >= 0) && (ww1 < W80);
        const uint16_t* Brow0 = x1b + (size_t)(hh0 * W80 + ww0) * C256 + c00;
        const uint16_t* Brow1 = x1b + (size_t)(hh1 * W80 + ww1) * C256 + c00;
        if (ok0) __builtin_prefetch(Brow0 + 32, 0, 0);
        if (ok1) __builtin_prefetch(Brow1 + 32, 0, 0);
        FragAB fa0, fa1, fb0, fb1;
#pragma unroll
        for (int v = 0; v < 8; ++v) {
            int ka = k0 + koffA(v, half), kb = koffB(v, half);
            fa0.u[v] = *(const uint32_t*)(Arow0 + ka);
            fa1.u[v] = *(const uint32_t*)(Arow1 + ka);
            fb0.u[v] = ok0 ? *(const uint32_t*)(Brow0 + kb) : 0u;
            fb1.u[v] = ok1 ? *(const uint32_t*)(Brow1 + kb) : 0u;
        }
        a00 = WMMA_BF16(fa0, fb0, a00);
        a01 = WMMA_BF16(fa0, fb1, a01);
        a10 = WMMA_BF16(fa1, fb0, a10);
        a11 = WMMA_BF16(fa1, fb1, a11);
    }
#pragma unroll
    for (int i = 0; i < 2; ++i) {
#pragma unroll
        for (int j = 0; j < 2; ++j) {
            const v8f& acc = i == 0 ? (j == 0 ? a00 : a01) : (j == 0 ? a10 : a11);
            int n = nbase + 16 * j + l16;
#pragma unroll
            for (int v = 0; v < 8; ++v) {
                int m = mbase + 16 * i + v + 8 * half;
                float xb = silu(acc[v] + b3[m] + b1[m]);
                size_t o = ((size_t)b * C256 + m) * P80 + n;
                out[o] = xb + x2[o];
            }
        }
    }
}

// ---------------------------------------------------------------------------
// Host launcher
// ---------------------------------------------------------------------------
extern "C" void kernel_launch(void* const* d_in, const int* in_sizes, int n_in,
                              void* d_out, int out_size, void* d_ws, size_t ws_size,
                              hipStream_t stream) {
    const float* y       = (const float*)d_in[0];
    const float* x       = (const float*)d_in[1];
    const float* nyg     = (const float*)d_in[2];
    const float* nyb     = (const float*)d_in[3];
    const float* nxg     = (const float*)d_in[4];
    const float* nxb     = (const float*)d_in[5];
    const float* q_w     = (const float*)d_in[6];
    const float* q_b     = (const float*)d_in[7];
    const float* k_w     = (const float*)d_in[8];
    const float* k_b     = (const float*)d_in[9];
    const float* cc_w    = (const float*)d_in[10];
    const float* cc_b    = (const float*)d_in[11];
    const float* ce_w    = (const float*)d_in[12];
    const float* ce_b    = (const float*)d_in[13];
    const float* f1_w    = (const float*)d_in[14];
    const float* f1_g    = (const float*)d_in[15];
    const float* f1_b    = (const float*)d_in[16];
    const float* f2_w    = (const float*)d_in[17];
    const float* f2_g    = (const float*)d_in[18];
    const float* f2_b    = (const float*)d_in[19];
    const float* r3_w    = (const float*)d_in[20];
    const float* r3_g    = (const float*)d_in[21];
    const float* r3_b    = (const float*)d_in[22];
    const float* r1_w    = (const float*)d_in[23];
    const float* r1_g    = (const float*)d_in[24];
    const float* r1_b    = (const float*)d_in[25];
    float* out = (float*)d_out;

    // workspace carve-out
    char* wp = (char*)d_ws;
    auto alloc = [&](size_t bytes) -> void* {
        void* p = (void*)wp;
        wp += (bytes + 255) & ~(size_t)255;
        return p;
    };
    float*    x_up   = (float*)   alloc((size_t)BATCH * C256 * P80 * 4);
    float*    kbuf   = (float*)   alloc((size_t)BATCH * P80 * EMB * 4);
    float*    ll     = (float*)   alloc((size_t)BATCH * C256 * P40 * 4);
    float*    yh     = (float*)   alloc((size_t)BATCH * C256 * P40 * 4);
    float*    q40    = (float*)   alloc((size_t)BATCH * P40 * EMB * 4);
    float*    com    = (float*)   alloc((size_t)BATCH * EMB * P40 * 4);
    float*    m100   = (float*)   alloc((size_t)BATCH * 100 * P40 * 4);
    uint16_t* cat    = (uint16_t*)alloc((size_t)BATCH * P80 * 512 * 2);
    uint16_t* x1     = (uint16_t*)alloc((size_t)BATCH * P80 * C256 * 2);
    float*    x2     = (float*)   alloc((size_t)BATCH * C256 * P80 * 4);
    uint16_t* Wf1    = (uint16_t*)alloc((size_t)C256 * KF * 2);
    uint16_t* Wf2    = (uint16_t*)alloc((size_t)C256 * KF * 2);
    uint16_t* Wr     = (uint16_t*)alloc((size_t)C256 * KR * 2);

    // 1) pack weights
    {
        int n = C256 * KR;
        pack_weights<<<(n + 255) / 256, 256, 0, stream>>>(f1_w, f1_g, f2_w, f2_g,
                                                          r3_w, r3_g, r1_w, r1_g,
                                                          Wf1, Wf2, Wr);
    }
    // 2) bilinear upsample
    {
        int n = BATCH * C256 * P80;
        upsample_x<<<(n + 255) / 256, 256, 0, stream>>>(x, x_up);
    }
    // 3) DWT
    {
        int n = BATCH * C256 * P40;
        dwt_haar<<<(n + 255) / 256, 256, 0, stream>>>(y, ll, yh);
    }
    // 4) layernorm + projections
    ln_proj<<<BATCH * P80, 64, 0, stream>>>(x_up, P80, nxg, nxb, k_w, k_b, kbuf);
    ln_proj<<<BATCH * P40, 64, 0, stream>>>(ll, P40, nyg, nyb, q_w, q_b, q40);
    // 5) attention -> cat[:,0:256]
    {
        int n = BATCH * P80;
        attn_kernel<<<(n + 255) / 256, 256, 0, stream>>>(q40, kbuf, x_up, cat);
    }
    // 6) HF mask path -> cat[:,256:512]
    {
        int n = BATCH * EMB * P40;
        com_kernel<<<(n + 255) / 256, 256, 0, stream>>>(yh, cc_w, cc_b, com);
        n = BATCH * 100 * P40;
        mask_conv_kernel<<<(n + 255) / 256, 256, 0, stream>>>(com, ce_w, ce_b, m100);
        n = BATCH * P80;
        hf_kernel<<<(n + 255) / 256, 256, 0, stream>>>(y, m100, cat);
    }
    // 7) WMMA GEMMs (64x64 block tile, 32x32 per wave)
    dim3 grid(P80 / 64, C256 / 64, BATCH);   // (100, 4, 2)
    gemm_f<<<grid, 128, 0, stream>>>(Wf1, cat, f1_b, x1, x2, 0);
    gemm_f<<<grid, 128, 0, stream>>>(Wf2, cat, f2_b, x1, x2, 1);
    gemm_r<<<grid, 128, 0, stream>>>(Wr, x1, r3_b, r1_b, x2, out);

    (void)in_sizes; (void)n_in; (void)out_size; (void)ws_size;
}
